// EquivariantDipoleMoment_42468636622964
// MI455X (gfx1250) — matrix operand order repaired and checked by reading, not compiled, not run on hardware
//
#include <hip/hip_runtime.h>

#define NATOMS 65536
#define NF     128
#define NBATCH 64
#define WPB    4          // waves per block
#define NTH    128        // threads per block
#define STRIPE (32 * 256) // elements per weight-strip buffer (max OUTS=256)

typedef unsigned int   u32;
typedef unsigned short u16;
typedef __attribute__((ext_vector_type(16))) __bf16 v16bf;
typedef __attribute__((ext_vector_type(8)))  __bf16 v8bf;
typedef __attribute__((ext_vector_type(8)))  float  v8f;
typedef __attribute__((ext_vector_type(8)))  u32    u32x8;
typedef __attribute__((ext_vector_type(4)))  u32    u32x4;
typedef __attribute__((ext_vector_type(2)))  u32    u32x2;

union Frag16 { v16bf v; u32x8 u; u32x4 q[2]; };
union BV8    { v8bf  b; u32x4 q; u16 s[8]; };

__device__ __forceinline__ u16 f2bf(float x) {
  union { __bf16 b; u16 s; } c; c.b = (__bf16)x; return c.s;
}
__device__ __forceinline__ float bf2f(u16 h) {
  union { u16 s; __bf16 b; } c; c.s = h; return (float)c.b;
}
__device__ __forceinline__ v8f v8f_zero() {
  v8f z;
#pragma unroll
  for (int i = 0; i < 8; ++i) z[i] = 0.f;
  return z;
}

// D-tile element r of acc maps to (out-feature = r + 8*(lane>>4), atom = lane&15).
__device__ __forceinline__ void store_tile(u16* dst, int stride, int col0,
                                           v8f a, int t, int lane) {
  const int hh = lane >> 4, m = lane & 15;
  union { u32x4 q; u16 s[8]; } pk;
#pragma unroll
  for (int r = 0; r < 8; ++r) pk.s[r] = f2bf(a[r]);
  *(u32x4*)(void*)(dst + m * stride + col0 + t * 16 + 8 * hh) = pk.q;
}

// Stage one transposed K-strip of W (f32->bf16) into LDS: strip[o][i], i=0..31.
// float4 over consecutive out-features keeps the global side coalesced.
template<int OUTS>
__device__ __forceinline__ void stage_strip(const float* __restrict__ W, int ks,
                                            u16* strip, int tid) {
  for (int idx = tid; idx < 8 * OUTS; idx += NTH) {
    const int i  = idx / (OUTS / 4);
    const int o4 = (idx % (OUTS / 4)) * 4;
    const float4 wv = *(const float4*)(W + (size_t)(ks * 32 + i) * OUTS + o4);
    strip[(o4 + 0) * 32 + i] = f2bf(wv.x);
    strip[(o4 + 1) * 32 + i] = f2bf(wv.y);
    strip[(o4 + 2) * 32 + i] = f2bf(wv.z);
    strip[(o4 + 3) * 32 + i] = f2bf(wv.w);
  }
  if (true) __builtin_prefetch(W + (size_t)(ks + 1) * 32 * OUTS, 0, 1);
}

// Generic D = W^T * X^T. Double-buffered weight strips: one barrier per strip,
// WMMA of strip k overlaps staging of strip k+1.
template<int INS, int OUTS, int BSTRIDE>
__device__ __forceinline__ void mm_bf16(const u16* actB, int bcol0,
                                        const float* __restrict__ W,
                                        u16* wbuf, v8f* acc,
                                        int lane, int tid) {
  const int hh = lane >> 4, m = lane & 15;
  for (int ks = 0; ks < INS / 32; ++ks) {
    u16* strip = wbuf + (ks & 1) * STRIPE;
    stage_strip<OUTS>(W, ks, strip, tid);
    __syncthreads();
    Frag16 b;
    b.u = *(const u32x8*)(const void*)(actB + m * BSTRIDE + bcol0 + ks * 32 + hh * 16);
#pragma unroll
    for (int t = 0; t < OUTS / 16; ++t) {
      const u16* ar = strip + (t * 16 + m) * 32 + 16 * hh;
      Frag16 a;
      a.q[0] = *(const u32x4*)(const void*)(ar);
      a.q[1] = *(const u32x4*)(const void*)(ar + 8);
      acc[t] = __builtin_amdgcn_wmma_f32_16x16x32_bf16(
          false, a.v, false, b.v, (short)0, acc[t], false, false);
    }
  }
}

// Batched variant: one strip feeds all 3 spatial dims (acc[3][OUTS/16]).
template<int OUTS, int BSTRIDE>
__device__ __forceinline__ void mm3_bf16(const u16* actB,
                                         const float* __restrict__ W,
                                         u16* wbuf, v8f* acc,
                                         int lane, int tid) {
  const int hh = lane >> 4, m = lane & 15;
  for (int ks = 0; ks < 4; ++ks) {                       // INS = 128
    u16* strip = wbuf + (ks & 1) * STRIPE;
    stage_strip<OUTS>(W, ks, strip, tid);
    __syncthreads();
#pragma unroll
    for (int d = 0; d < 3; ++d) {
      Frag16 b;
      b.u = *(const u32x8*)(const void*)(actB + m * BSTRIDE + d * 128 + ks * 32 + hh * 16);
#pragma unroll
      for (int t = 0; t < OUTS / 16; ++t) {
        const u16* ar = strip + (t * 16 + m) * 32 + 16 * hh;
        Frag16 a;
        a.q[0] = *(const u32x4*)(const void*)(ar);
        a.q[1] = *(const u32x4*)(const void*)(ar + 8);
        acc[d * (OUTS / 16) + t] = __builtin_amdgcn_wmma_f32_16x16x32_bf16(
            false, a.v, false, b.v, (short)0, acc[d * (OUTS / 16) + t], false, false);
      }
    }
  }
}

// One GatedEquivariant block; act buffers per-wave, weight strips block-shared.
__device__ __forceinline__ void gated_block(
    u16* actH, u16* actV, u16* v2buf, u16* wbuf,
    const float* Wu, const float* Wv,
    const float* W1, const float* b1,
    const float* W2, const float* b2, int lane, int tid) {
  const int hh = lane >> 4, m = lane & 15;

  // v1 = v @ Wu (all 3 dims per strip)
  v8f va[24];
#pragma unroll
  for (int i = 0; i < 24; ++i) va[i] = v8f_zero();
  mm3_bf16<128, 384>(actV, Wu, wbuf, va, lane, tid);

  // vn = sqrt(sum_d v1_d^2) -> actH cols 128..255
#pragma unroll
  for (int t = 0; t < 8; ++t) {
    v8f vn = va[t] * va[t] + va[8 + t] * va[8 + t] + va[16 + t] * va[16 + t];
    v8f sq;
#pragma unroll
    for (int r = 0; r < 8; ++r) sq[r] = sqrtf(vn[r]);
    store_tile(actH, 256, 128, sq, t, lane);
  }

  // v2 = v @ Wv -> v2buf
#pragma unroll
  for (int i = 0; i < 24; ++i) va[i] = v8f_zero();
  mm3_bf16<128, 384>(actV, Wv, wbuf, va, lane, tid);
#pragma unroll
  for (int d = 0; d < 3; ++d)
#pragma unroll
    for (int t = 0; t < 8; ++t)
      store_tile(v2buf, 384, d * 128, va[d * 8 + t], t, lane);

  // h1 = silu([s, vn] @ W1 + b1) -> actV cols 0..127 (old v dead)
  v8f h[8];
#pragma unroll
  for (int t = 0; t < 8; ++t) h[t] = v8f_zero();
  mm_bf16<256, 128, 256>(actH, 0, W1, wbuf, h, lane, tid);
#pragma unroll
  for (int t = 0; t < 8; ++t) {
#pragma unroll
    for (int r = 0; r < 8; ++r) {
      const int o = t * 16 + 8 * hh + r;
      float x = h[t][r] + b1[o];
      h[t][r] = x / (1.f + __expf(-x));
    }
    store_tile(actV, 384, 0, h[t], t, lane);
  }

  // h2 = h1 @ W2 + b2 ; tiles 0..7 -> new s, tiles 8..15 -> gate
  v8f g[16];
#pragma unroll
  for (int t = 0; t < 16; ++t) g[t] = v8f_zero();
  mm_bf16<128, 256, 384>(actV, 0, W2, wbuf, g, lane, tid);

#pragma unroll
  for (int t = 0; t < 8; ++t) {
    v8f so;
#pragma unroll
    for (int r = 0; r < 8; ++r) so[r] = g[t][r] + b2[t * 16 + 8 * hh + r];
    store_tile(actH, 256, 0, so, t, lane);
  }
  // gate: 8 contiguous features per (t,lane); bf16 vector multiply (v_pk_mul_bf16)
#pragma unroll
  for (int t = 8; t < 16; ++t) {
    const int fb = (t - 8) * 16 + 8 * hh;                // 8-aligned feature base
    BV8 gv;
#pragma unroll
    for (int r = 0; r < 8; ++r) {
      union { __bf16 b; u16 s; } c;
      c.b = (__bf16)(g[t][r] + b2[128 + fb + r]);
      gv.s[r] = c.s;
    }
#pragma unroll
    for (int d = 0; d < 3; ++d) {
      BV8 vv, nv;
      vv.q = *(const u32x4*)(const void*)(v2buf + m * 384 + d * 128 + fb);
      nv.b = vv.b * gv.b;
      *(u32x4*)(void*)(actV + m * 384 + d * 128 + fb) = nv.q;
    }
  }
}

__global__ void zero_ws_kernel(float* wsf) {
  const int i = threadIdx.x;
  if (i < 512) wsf[i] = 0.f;
}

__global__ void centre_kernel(const float* __restrict__ z,
                              const float* __restrict__ r,
                              const float* __restrict__ mask,
                              int* __restrict__ ids,
                              float* __restrict__ cnum,
                              float* __restrict__ cden) {
  const int n = blockIdx.x * blockDim.x + threadIdx.x;
  if (n >= NATOMS) return;
  int id = 0;
  for (int b = 0; b < NBATCH; ++b)
    if (mask[b * NATOMS + n] > 0.5f) id = b;
  ids[n] = id;
  const float zz = z[n];
  atomicAdd(&cden[id], zz);
#pragma unroll
  for (int d = 0; d < 3; ++d)
    atomicAdd(&cnum[id * 3 + d], zz * r[n * 3 + d]);
}

__global__ __launch_bounds__(NTH)
void eqdm_main_kernel(const float* __restrict__ s, const float* __restrict__ v,
                      const float* __restrict__ r,
                      const float* Wu0, const float* Wv0, const float* W1_0,
                      const float* b1_0, const float* W2_0, const float* b2_0,
                      const float* Wu1, const float* Wv1, const float* W1_1,
                      const float* b1_1, const float* W2_1, const float* b2_1,
                      const float* __restrict__ Ws, const float* __restrict__ Wvf,
                      const int* __restrict__ ids,
                      float* __restrict__ mu, float* __restrict__ stot) {
  __shared__ __align__(32) u16 wbuf[2 * STRIPE];          // 2 x 16 KB ping-pong
  __shared__ __align__(32) u16 actH[WPB][16 * 256];       // 4 x 8 KB
  __shared__ __align__(32) u16 actV[WPB][16 * 384];       // 4 x 12 KB
  __shared__ __align__(32) u16 v2buf[WPB][16 * 384];      // 4 x 12 KB

  const int tid  = threadIdx.x;
  const int w    = tid >> 5;
  const int lane = tid & 31;
  const int base = (blockIdx.x * WPB + w) * 16;

  u16* aH = actH[w];
  u16* aV = actV[w];
  u16* v2 = v2buf[w];

  // stage activations f32 -> bf16, 16B global loads + 8B LDS stores
  for (int a = 0; a < 16; ++a) {
    const float4 sv = *(const float4*)(s + (size_t)(base + a) * NF + lane * 4);
    union { u32x2 q; u16 p[4]; } pk;
    pk.p[0] = f2bf(sv.x); pk.p[1] = f2bf(sv.y);
    pk.p[2] = f2bf(sv.z); pk.p[3] = f2bf(sv.w);
    *(u32x2*)(void*)(aH + a * 256 + lane * 4) = pk.q;
  }
  for (int a = 0; a < 16; ++a) {
#pragma unroll
    for (int k = 0; k < 3; ++k) {
      const float4 vv = *(const float4*)(v + (size_t)(base + a) * 384 + k * 128 + lane * 4);
      union { u32x2 q; u16 p[4]; } pk;
      pk.p[0] = f2bf(vv.x); pk.p[1] = f2bf(vv.y);
      pk.p[2] = f2bf(vv.z); pk.p[3] = f2bf(vv.w);
      *(u32x2*)(void*)(aV + a * 384 + k * 128 + lane * 4) = pk.q;
    }
  }
  // per-wave DS ordering covers our own slab; staging barriers cover wbuf

  gated_block(aH, aV, v2, wbuf, Wu0, Wv0, W1_0, b1_0, W2_0, b2_0, lane, tid);
  gated_block(aH, aV, v2, wbuf, Wu1, Wv1, W1_1, b1_1, W2_1, b2_1, lane, tid);

  // final projections: 8-wide bf16 LDS reads, fold halves with shfl_xor(16)
  const int a  = lane & 15;
  const int hs = lane >> 4;
  float sdot = 0.f;
#pragma unroll
  for (int j = 0; j < 64; j += 8) {
    const int f = hs * 64 + j;
    BV8 hv; hv.q = *(const u32x4*)(const void*)(aH + a * 256 + f);
#pragma unroll
    for (int rr = 0; rr < 8; ++rr) sdot += bf2f(hv.s[rr]) * Ws[f + rr];
  }
  float vd[3];
#pragma unroll
  for (int d = 0; d < 3; ++d) {
    float x = 0.f;
#pragma unroll
    for (int j = 0; j < 64; j += 8) {
      const int f = hs * 64 + j;
      BV8 hv; hv.q = *(const u32x4*)(const void*)(aV + a * 384 + d * 128 + f);
#pragma unroll
      for (int rr = 0; rr < 8; ++rr) x += bf2f(hv.s[rr]) * Wvf[f + rr];
    }
    vd[d] = x;
  }
  sdot += __shfl_xor(sdot, 16, 32);
#pragma unroll
  for (int d = 0; d < 3; ++d) vd[d] += __shfl_xor(vd[d], 16, 32);

  if (lane < 16) {
    const int id = ids[base + a];
    atomicAdd(stot, sdot);
#pragma unroll
    for (int d = 0; d < 3; ++d) {
      const float contrib = vd[d] + sdot * r[(base + a) * 3 + d];
      atomicAdd(&mu[id * 3 + d], contrib);
    }
  }
}

__global__ void finalize_kernel(const float* __restrict__ mu,
                                const float* __restrict__ cnum,
                                const float* __restrict__ cden,
                                const float* __restrict__ stot,
                                float* __restrict__ out) {
  const int b = threadIdx.x;
  if (b >= NBATCH) return;
  const float st = *stot;
  const float den = cden[b];
  float acc = 0.f;
#pragma unroll
  for (int d = 0; d < 3; ++d) {
    const float c = cnum[b * 3 + d] / den;
    const float m = mu[b * 3 + d] - st * c;
    acc += m * m;
  }
  out[b] = sqrtf(acc);
}

extern "C" void kernel_launch(void* const* d_in, const int* in_sizes, int n_in,
                              void* d_out, int out_size, void* d_ws, size_t ws_size,
                              hipStream_t stream) {
  (void)in_sizes; (void)n_in; (void)out_size; (void)ws_size;
  const float* z    = (const float*)d_in[0];
  const float* s    = (const float*)d_in[1];
  const float* v    = (const float*)d_in[2];
  const float* r    = (const float*)d_in[3];
  const float* mask = (const float*)d_in[4];
  const float* Wu0  = (const float*)d_in[5];
  const float* Wv0  = (const float*)d_in[6];
  const float* W1_0 = (const float*)d_in[7];
  const float* b1_0 = (const float*)d_in[8];
  const float* W2_0 = (const float*)d_in[9];
  const float* b2_0 = (const float*)d_in[10];
  const float* Wu1  = (const float*)d_in[11];
  const float* Wv1  = (const float*)d_in[12];
  const float* W1_1 = (const float*)d_in[13];
  const float* b1_1 = (const float*)d_in[14];
  const float* W2_1 = (const float*)d_in[15];
  const float* b2_1 = (const float*)d_in[16];
  const float* Ws   = (const float*)d_in[17];
  const float* Wvf  = (const float*)d_in[18];

  float* wsf  = (float*)d_ws;
  float* mu   = wsf;                // [64*3]
  float* cnum = wsf + 192;          // [64*3]
  float* cden = wsf + 384;          // [64]
  float* stot = wsf + 448;          // [1]
  int*   ids  = (int*)(wsf + 512);  // [65536]

  zero_ws_kernel<<<1, 512, 0, stream>>>(wsf);
  centre_kernel<<<NATOMS / 256, 256, 0, stream>>>(z, r, mask, ids, cnum, cden);
  eqdm_main_kernel<<<NATOMS / (16 * WPB), NTH, 0, stream>>>(
      s, v, r,
      Wu0, Wv0, W1_0, b1_0, W2_0, b2_0,
      Wu1, Wv1, W1_1, b1_1, W2_1, b2_1,
      Ws, Wvf, ids, mu, stot);
  finalize_kernel<<<1, NBATCH, 0, stream>>>(mu, cnum, cden, stot, (float*)d_out);
}
